// Custom_criterion_4561255268904
// MI455X (gfx1250) — matrix operand order repaired and checked by reading, not compiled, not run on hardware
//
#include <hip/hip_runtime.h>

typedef __attribute__((ext_vector_type(16))) _Float16 v16h;
typedef __attribute__((ext_vector_type(8)))  _Float16 v8h;
typedef __attribute__((ext_vector_type(8)))  float    v8f;
typedef __attribute__((ext_vector_type(4)))  float    v4f;

#define IMG_W        512
#define CH_ELEMS     (512 * 512)
#define GRID_MAIN    3072
#define WAVES_PB     8
#define TILES_PW     4          // 3072 * 8 * 4 = 98304 = 96 * 32 * 32 tiles
#define C1F          6.5025f
#define C2F          58.5225f
#define MSE_COUNT    25165824.0f   // 32*3*512*512
#define SSIM_COUNT   23712864.0f   // 96*497*497

__device__ __forceinline__ float gaussw(int i) {
  float d = (float)(i - 8);
  return __expf(-d * d * (1.0f / 4.5f));   // 2*sigma^2 = 4.5
}

__device__ __forceinline__ v8h cvt8(v8f d) {
  v8h p;
#pragma unroll
  for (int i = 0; i < 8; ++i) p[i] = (_Float16)d[i];
  return p;
}

__global__ __launch_bounds__(256)
void ssim_mse_main(const float* __restrict__ gout, const float* __restrict__ gtg,
                   float* __restrict__ partial)
{
  // Per-wave transposed H buffer: [wave][map][n=col][k/8], pitch 40 halves (80B, conflict-free)
  __shared__ v8h   hbuf[WAVES_PB][5][16][5];
  __shared__ float red[512];

  const int tid    = threadIdx.x;
  const int wave   = tid >> 5;
  const int lane   = tid & 31;
  const int laneHi = lane >> 4;   // 0 or 1
  const int lrow   = lane & 15;

  // Gaussian normalization
  float gs = 0.f;
#pragma unroll
  for (int i = 0; i < 16; ++i) gs += gaussw(i);
  const float ginv = 1.0f / gs;

  // Toeplitz tap matrix T(32x16), T[k][n] = g[k-n], in WMMA B layout:
  // element e holds K = e + 8*laneHi (e<8) or e + 8 + 8*laneHi (e>=8), column N = lrow.
  v16h bmat;
#pragma unroll
  for (int e = 0; e < 16; ++e) {
    const int k = (e < 8) ? (e + 8 * laneHi) : (e + 8 + 8 * laneHi);
    const int d = k - lrow;
    float v = 0.f;
    if (d >= 0 && d < 16) v = gaussw(d) * ginv;
    bmat[e] = (_Float16)v;
  }

  float ssim_acc = 0.f;
  float mse_acc  = 0.f;

  for (int t = 0; t < TILES_PW; ++t) {
    const int tileIdx = (blockIdx.x * WAVES_PB + wave) + t * (GRID_MAIN * WAVES_PB);
    const int ch  = tileIdx >> 10;          // 1024 tiles per channel-image
    const int rem = tileIdx & 1023;
    const int r0  = (rem >> 5) << 4;
    const int c0  = (rem & 31) << 4;
    const float* xc = gout + (size_t)ch * CH_ELEMS;
    const float* yc = gtg  + (size_t)ch * CH_ELEMS;

    // ---- horizontal pass: two 16-row blocks -> transposed f16 H in LDS ----
#pragma unroll
    for (int rb = 0; rb < 2; ++rb) {
      int row = r0 + rb * 16 + lrow;   if (row > 511) row = 511;
      int cb0 = c0 + 8 * laneHi;       if (cb0 > 504) cb0 = 504;
      int cb1 = c0 + 16 + 8 * laneHi;  if (cb1 > 504) cb1 = 504;
      const float* xr = xc + row * IMG_W;
      const float* yr = yc + row * IMG_W;
      v4f x0 = *(const v4f*)(xr + cb0);
      v4f x1 = *(const v4f*)(xr + cb0 + 4);
      v4f x2 = *(const v4f*)(xr + cb1);
      v4f x3 = *(const v4f*)(xr + cb1 + 4);
      v4f y0 = *(const v4f*)(yr + cb0);
      v4f y1 = *(const v4f*)(yr + cb0 + 4);
      v4f y2 = *(const v4f*)(yr + cb1);
      v4f y3 = *(const v4f*)(yr + cb1 + 4);

      float xv[16], yv[16];
#pragma unroll
      for (int i = 0; i < 4; ++i) {
        xv[i] = x0[i]; xv[4 + i] = x1[i]; xv[8 + i] = x2[i]; xv[12 + i] = x3[i];
        yv[i] = y0[i]; yv[4 + i] = y1[i]; yv[8 + i] = y2[i]; yv[12 + i] = y3[i];
      }

      // MSE: top block, first 8 elems per lane = exact disjoint 512x512 cover
      if (rb == 0) {
#pragma unroll
        for (int i = 0; i < 8; ++i) { float d = xv[i] - yv[i]; mse_acc += d * d; }
      }

      // A operands (WMMA A layout: e<8 -> K=e+8*laneHi(chunk0), e>=8 -> K=e+8+8*laneHi)
      v16h ax, ay, axx, ayy, axy;
#pragma unroll
      for (int e = 0; e < 16; ++e) {
        ax[e]  = (_Float16)xv[e];
        ay[e]  = (_Float16)yv[e];
        axx[e] = (_Float16)(xv[e] * xv[e]);
        ayy[e] = (_Float16)(yv[e] * yv[e]);
        axy[e] = (_Float16)(xv[e] * yv[e]);
      }

      v8f cz = {};
      v8f d0 = __builtin_amdgcn_wmma_f32_16x16x32_f16(false, ax,  false, bmat, (short)0, cz, false, false);
      v8f d1 = __builtin_amdgcn_wmma_f32_16x16x32_f16(false, ay,  false, bmat, (short)0, cz, false, false);
      v8f d2 = __builtin_amdgcn_wmma_f32_16x16x32_f16(false, axx, false, bmat, (short)0, cz, false, false);
      v8f d3 = __builtin_amdgcn_wmma_f32_16x16x32_f16(false, ayy, false, bmat, (short)0, cz, false, false);
      v8f d4 = __builtin_amdgcn_wmma_f32_16x16x32_f16(false, axy, false, bmat, (short)0, cz, false, false);

      const int slot = rb * 2 + laneHi;     // = (rb*16 + 8*laneHi)/8
      hbuf[wave][0][lrow][slot] = cvt8(d0);
      hbuf[wave][1][lrow][slot] = cvt8(d1);
      hbuf[wave][2][lrow][slot] = cvt8(d2);
      hbuf[wave][3][lrow][slot] = cvt8(d3);
      hbuf[wave][4][lrow][slot] = cvt8(d4);
    }
    __syncthreads();

    // ---- vertical pass: D = H^T (16x32) x Toeplitz(32x16) ----
    v8f D[5];
#pragma unroll
    for (int m = 0; m < 5; ++m) {
      v8h a0 = hbuf[wave][m][lrow][laneHi];        // K = 8*laneHi .. +7
      v8h a1 = hbuf[wave][m][lrow][2 + laneHi];    // K = 16+8*laneHi .. +7
      v16h a;
#pragma unroll
      for (int e = 0; e < 8; ++e) { a[e] = a0[e]; a[e + 8] = a1[e]; }
      v8f cz = {};
      D[m] = __builtin_amdgcn_wmma_f32_16x16x32_f16(false, a, false, bmat, (short)0, cz, false, false);
    }

    // ---- SSIM map + masked reduce (output tile is transposed: row<->col) ----
    const int orow = r0 + lrow;
#pragma unroll
    for (int v = 0; v < 8; ++v) {
      const int ocol = c0 + v + 8 * laneHi;
      float mu1 = D[0][v], mu2 = D[1][v];
      float mu1s = mu1 * mu1, mu2s = mu2 * mu2, mu12 = mu1 * mu2;
      float s1  = D[2][v] - mu1s;
      float s2  = D[3][v] - mu2s;
      float s12 = D[4][v] - mu12;
      float num = (2.f * mu12 + C1F) * (2.f * s12 + C2F);
      float den = (mu1s + mu2s + C1F) * (s1 + s2 + C2F);
      float sv  = num / den;
      ssim_acc += (orow < 497 && ocol < 497) ? sv : 0.f;
    }
    __syncthreads();
  }

  // ---- deterministic block reduction ----
  red[tid] = ssim_acc;
  red[256 + tid] = mse_acc;
  __syncthreads();
  for (int s = 128; s > 0; s >>= 1) {
    if (tid < s) { red[tid] += red[tid + s]; red[256 + tid] += red[256 + tid + s]; }
    __syncthreads();
  }
  if (tid == 0) {
    partial[2 * blockIdx.x]     = red[0];
    partial[2 * blockIdx.x + 1] = red[256];
  }
}

__global__ __launch_bounds__(256)
void ssim_mse_finalize(const float* __restrict__ partial, float* __restrict__ out)
{
  __shared__ float red[512];
  float s = 0.f, m = 0.f;
  for (int i = threadIdx.x; i < GRID_MAIN; i += 256) {
    s += partial[2 * i];
    m += partial[2 * i + 1];
  }
  red[threadIdx.x] = s;
  red[256 + threadIdx.x] = m;
  __syncthreads();
  for (int st = 128; st > 0; st >>= 1) {
    if (threadIdx.x < st) {
      red[threadIdx.x] += red[threadIdx.x + st];
      red[256 + threadIdx.x] += red[256 + threadIdx.x + st];
    }
    __syncthreads();
  }
  if (threadIdx.x == 0) {
    const float mse  = red[256] * (1.0f / MSE_COUNT);
    const float ssim = red[0]   * (1.0f / SSIM_COUNT);
    out[0] = mse + (1.0f - ssim);
  }
}

extern "C" void kernel_launch(void* const* d_in, const int* in_sizes, int n_in,
                              void* d_out, int out_size, void* d_ws, size_t ws_size,
                              hipStream_t stream)
{
  (void)in_sizes; (void)n_in; (void)out_size; (void)ws_size;
  const float* gout = (const float*)d_in[0];   // "output" image batch, fp32
  const float* gtg  = (const float*)d_in[1];   // "target" image batch, fp32
  float* partial = (float*)d_ws;               // 3072 * 2 floats = 24 KB

  ssim_mse_main<<<GRID_MAIN, 256, 0, stream>>>(gout, gtg, partial);
  ssim_mse_finalize<<<1, 256, 0, stream>>>(partial, (float*)d_out);
}
